// TurboQuantMoEExpert_9689446219886
// MI455X (gfx1250) — compile-verified, matrix operand order checked by reading, not run on hardware
//
#include <hip/hip_runtime.h>
#include <hip/hip_bf16.h>

// TurboQuantMoEExpert for MI455X (gfx1250, wave32, WMMA).
//
// Pipeline (all on `stream`):
//   K0 x3 : dequantize 4-bit packed weights -> scaled f16 (codebook LUT * norms/sqrt(128))
//   K1 x2 : rotate x by per-group orthogonal matrices (f16 WMMA) -> Xg, Xu
//   K2    : fused gate/up GEMM (f16 WMMA, f32 acc, LDS-staged weights) + SwiGLU -> H
//   K1'   : rotate H by rot_down -> Hrot
//   K3    : down GEMM (LDS-staged weights) + bias -> f32 output
//
// Roofline: ~106 GFLOP vs ~100MB unique bytes => bound by the WMMA pipe, not HBM.
// Pre-dequantized f16 weights (96MB) are L2-resident (192MB L2). Weight tiles are
// staged once per block in LDS (double-buffered) so the 8 waves of a block do not
// each re-fetch the same fragments from L2; the stage uses the CDNA5 async
// global->LDS path (GLOBAL_LOAD_ASYNC_TO_LDS_B128, ASYNCcnt).

#define GRP       128
#define IN_DIM    2048
#define INTER_DIM 8192
#define BATCH     1024
#define INV_SQRT_G 0.08838834764831845f   // 1/sqrt(128)

typedef __attribute__((ext_vector_type(16))) _Float16 v16h;
typedef __attribute__((ext_vector_type(8)))  _Float16 v8h;
typedef __attribute__((ext_vector_type(8)))  float    v8f;
typedef int v4i __attribute__((vector_size(16)));

// Async global->LDS copy (GLOBAL_LOAD_ASYNC_TO_LDS_B128, ASYNCcnt) if declared.
#if defined(__has_builtin)
#if __has_builtin(__builtin_amdgcn_global_load_async_to_lds_b128) && \
    __has_builtin(__builtin_amdgcn_s_wait_asynccnt)
#define USE_ASYNC_COPY 1
#endif
#endif

#if defined(USE_ASYNC_COPY)
// Probe-derived signature: param 1 is AS1 (global, "__device__") v4i*; dst is LDS.
typedef __attribute__((address_space(1))) v4i* gptr_t;
typedef __attribute__((address_space(3))) v4i* lptr_t;
static __device__ __forceinline__ void async_copy16(const void* g, void* l) {
  __builtin_amdgcn_global_load_async_to_lds_b128(
      (gptr_t)g, (lptr_t)l, /*imm offset=*/0, /*cpol=*/0);
}
#endif

// ---------------------------------------------------------------------------
// Fragment loaders (16x32 f16 A-layout; symmetric layout used for B = W[N,K]).
// lane 0-15 : row = lane,     K = {0..7, 16..23}
// lane16-31 : row = lane-16,  K = {8..15, 24..31}
// ---------------------------------------------------------------------------
static __device__ __forceinline__ v16h load_frag(const _Float16* __restrict__ base,
                                                 int ld, int lane) {
  const int r  = lane & 15;
  const int ko = (lane >> 4) << 3;           // 0 or 8
  const _Float16* p = base + (size_t)r * ld + ko;
  v8h lo = *(const v8h*)(p);
  v8h hi = *(const v8h*)(p + 16);
  return __builtin_shufflevector(lo, hi, 0,1,2,3,4,5,6,7,8,9,10,11,12,13,14,15);
}

static __device__ __forceinline__ v16h load_frag(const float* __restrict__ base,
                                                 int ld, int lane) {
  const int r  = lane & 15;
  const int ko = (lane >> 4) << 3;
  const float* p = base + (size_t)r * ld + ko;
  v8f lo = *(const v8f*)(p);
  v8f hi = *(const v8f*)(p + 16);
  v8h l2 = __builtin_convertvector(lo, v8h);
  v8h h2 = __builtin_convertvector(hi, v8h);
  return __builtin_shufflevector(l2, h2, 0,1,2,3,4,5,6,7,8,9,10,11,12,13,14,15);
}

// B-fragment from an LDS tile [rows][32] f16, row-major, base at fragment row 0.
static __device__ __forceinline__ v16h load_frag_lds(const _Float16* base, int lane) {
  const int r  = lane & 15;
  const int ko = (lane >> 4) << 3;
  const _Float16* p = base + r * 32 + ko;
  v8h lo = *(const v8h*)(p);                 // ds_load_b128
  v8h hi = *(const v8h*)(p + 16);
  return __builtin_shufflevector(lo, hi, 0,1,2,3,4,5,6,7,8,9,10,11,12,13,14,15);
}

static __device__ __forceinline__ v8f wmma_f16(v16h a, v16h b, v8f c) {
  return __builtin_amdgcn_wmma_f32_16x16x32_f16(
      false, a, false, b, (short)0, c, false, false);
}

// ---------------------------------------------------------------------------
// K0: dequantize packed 4-bit codes -> scaled f16 weights.
// ---------------------------------------------------------------------------
__global__ __launch_bounds__(256) void dequant_kernel(
    const int*   __restrict__ packed,   // [N, K/2] one byte per int32
    const float* __restrict__ codebook, // [16]
    const float* __restrict__ norms,    // [N, K/128]
    _Float16*    __restrict__ Wout,     // [N, K] f16 row-major
    int N, int K) {
  __shared__ float cb[16];
  if (threadIdx.x < 16) cb[threadIdx.x] = codebook[threadIdx.x];
  __syncthreads();

  const int half_k = K >> 1;
  const int idx = blockIdx.x * blockDim.x + threadIdx.x;
  if (idx >= N * half_k) return;
  const int n = idx / half_k;
  const int p = idx - n * half_k;
  const int k = p << 1;
  const float s = norms[(size_t)n * (K >> 7) + (k >> 7)] * INV_SQRT_G;
  const int byte = packed[idx];
  _Float16* dst = Wout + (size_t)n * K + k;
  dst[0] = (_Float16)(cb[byte & 15] * s);
  dst[1] = (_Float16)(cb[(byte >> 4) & 15] * s);
}

// ---------------------------------------------------------------------------
// K1: per-group rotation: Xrot[b, g*128+j] = sum_i X[b, g*128+i] * rot[g,j,i]
//     grid = (B/128, n_groups), block = 256 (8 waves); each wave: 16(M)x128(N)
// ---------------------------------------------------------------------------
template <typename T>
__global__ __launch_bounds__(256) void rotate_kernel(
    const T*     __restrict__ X,    // [B, width]
    const float* __restrict__ rot,  // [ng, 128, 128] row j, col i
    _Float16*    __restrict__ Xrot, // [B, width] f16
    int width) {
  const int lane = threadIdx.x & 31;
  const int wave = threadIdx.x >> 5;
  const int g    = blockIdx.y;
  const int row0 = blockIdx.x * 128 + wave * 16;

  const float* R     = rot + (size_t)g * GRP * GRP;
  const T*     Xbase = X + (size_t)row0 * width + g * GRP;

  v8f acc[8] = {};
  #pragma unroll
  for (int kk = 0; kk < GRP; kk += 32) {
    v16h a = load_frag(Xbase + kk, width, lane);
    #pragma unroll
    for (int t = 0; t < 8; ++t) {
      v16h b = load_frag(R + (size_t)(t * 16) * GRP + kk, GRP, lane);
      acc[t] = wmma_f16(a, b, acc[t]);
    }
  }

  const int rbase = (lane >> 4) << 3;
  const int c     = lane & 15;
  #pragma unroll
  for (int t = 0; t < 8; ++t) {
    #pragma unroll
    for (int i = 0; i < 8; ++i) {
      Xrot[(size_t)(row0 + rbase + i) * width + g * GRP + t * 16 + c] =
          (_Float16)acc[t][i];
    }
  }
}

// ---------------------------------------------------------------------------
// K2: fused gate/up GEMM + bias + SwiGLU -> H f16
//     grid = (B/128, INTER/64), block 256 (8 waves), wave tile 16x64 per matrix.
//     Weight tiles (64x32 f16 = 4KB each) staged in LDS, double-buffered.
// ---------------------------------------------------------------------------
__global__ __launch_bounds__(256) void gateup_kernel(
    const _Float16* __restrict__ Xg, const _Float16* __restrict__ Xu,   // [B,K]
    const _Float16* __restrict__ Wg, const _Float16* __restrict__ Wu,   // [N,K]
    const float* __restrict__ gate_bias, const float* __restrict__ up_bias,
    _Float16* __restrict__ H, int K, int N) {
  __shared__ _Float16 sWg[2][64 * 32];
  __shared__ _Float16 sWu[2][64 * 32];

  const int tid  = threadIdx.x;
  const int lane = tid & 31;
  const int wave = tid >> 5;
  const int row0 = blockIdx.x * 128 + wave * 16;
  const int col0 = blockIdx.y * 64;

  // Cooperative stage mapping: thread copies one b128 (8 halves) per tile.
  const int cn = tid >> 2;               // 0..63  (local n)
  const int ck = (tid & 3) << 3;         // 0,8,16,24 (local k)
  const _Float16* gWg = Wg + (size_t)(col0 + cn) * K + ck;
  const _Float16* gWu = Wu + (size_t)(col0 + cn) * K + ck;

  const _Float16* Ag = Xg + (size_t)row0 * K;
  const _Float16* Au = Xu + (size_t)row0 * K;

  v8f accg[4] = {};
  v8f accu[4] = {};
  const int NK = K / 32;

#if defined(USE_ASYNC_COPY)
  async_copy16(gWg, &sWg[0][cn * 32 + ck]);
  async_copy16(gWu, &sWu[0][cn * 32 + ck]);
#else
  v8h rg = *(const v8h*)(gWg);
  v8h ru = *(const v8h*)(gWu);
#endif

  for (int k = 0; k < NK - 1; ++k) {
    const int cur = k & 1, nxt = cur ^ 1;
    const int kk = k * 32;
#if defined(USE_ASYNC_COPY)
    async_copy16(gWg + (size_t)(k + 1) * 32, &sWg[nxt][cn * 32 + ck]);
    async_copy16(gWu + (size_t)(k + 1) * 32, &sWu[nxt][cn * 32 + ck]);
    __builtin_amdgcn_s_wait_asynccnt(2);     // tile k landed (in-order)
    __syncthreads();
#else
    *(v8h*)(&sWg[cur][cn * 32 + ck]) = rg;   // ds_store_b128
    *(v8h*)(&sWu[cur][cn * 32 + ck]) = ru;
    __syncthreads();
    rg = *(const v8h*)(gWg + (size_t)(k + 1) * 32);  // overlap with WMMAs below
    ru = *(const v8h*)(gWu + (size_t)(k + 1) * 32);
#endif
    __builtin_prefetch(Ag + kk + 128, 0, 3);  // global_prefetch_b8
    __builtin_prefetch(Au + kk + 128, 0, 3);
    v16h ag = load_frag(Ag + kk, K, lane);
    v16h au = load_frag(Au + kk, K, lane);
    #pragma unroll
    for (int t = 0; t < 4; ++t) {
      v16h bg = load_frag_lds(&sWg[cur][(t * 16) * 32], lane);
      v16h bu = load_frag_lds(&sWu[cur][(t * 16) * 32], lane);
      accg[t] = wmma_f16(ag, bg, accg[t]);
      accu[t] = wmma_f16(au, bu, accu[t]);
    }
    __syncthreads();
  }
  {  // epilogue tile NK-1
    const int cur = (NK - 1) & 1;
    const int kk = (NK - 1) * 32;
#if defined(USE_ASYNC_COPY)
    __builtin_amdgcn_s_wait_asynccnt(0);
    __syncthreads();
#else
    *(v8h*)(&sWg[cur][cn * 32 + ck]) = rg;
    *(v8h*)(&sWu[cur][cn * 32 + ck]) = ru;
    __syncthreads();
#endif
    v16h ag = load_frag(Ag + kk, K, lane);
    v16h au = load_frag(Au + kk, K, lane);
    #pragma unroll
    for (int t = 0; t < 4; ++t) {
      v16h bg = load_frag_lds(&sWg[cur][(t * 16) * 32], lane);
      v16h bu = load_frag_lds(&sWu[cur][(t * 16) * 32], lane);
      accg[t] = wmma_f16(ag, bg, accg[t]);
      accu[t] = wmma_f16(au, bu, accu[t]);
    }
  }

  const int rbase = (lane >> 4) << 3;
  const int c     = lane & 15;
  #pragma unroll
  for (int t = 0; t < 4; ++t) {
    #pragma unroll
    for (int i = 0; i < 8; ++i) {
      const int n  = col0 + t * 16 + c;
      const float g = accg[t][i] + gate_bias[n];
      const float u = accu[t][i] + up_bias[n];
      const float h = (g / (1.0f + __expf(-g))) * u;   // silu(g) * u
      H[(size_t)(row0 + rbase + i) * N + n] = (_Float16)h;
    }
  }
}

// ---------------------------------------------------------------------------
// K3: down GEMM + bias -> f32 output. Same LDS staging, single weight matrix.
//     grid = (B/128, IN/64), block 256.
// ---------------------------------------------------------------------------
__global__ __launch_bounds__(256) void down_kernel(
    const _Float16* __restrict__ Hrot, // [B, K=8192]
    const _Float16* __restrict__ Wd,   // [N=2048, K]
    const float*    __restrict__ bias, // [N]
    float*          __restrict__ out,  // [B, N] f32
    int K, int N) {
  __shared__ _Float16 sW[2][64 * 32];

  const int tid  = threadIdx.x;
  const int lane = tid & 31;
  const int wave = tid >> 5;
  const int row0 = blockIdx.x * 128 + wave * 16;
  const int col0 = blockIdx.y * 64;

  const int cn = tid >> 2;
  const int ck = (tid & 3) << 3;
  const _Float16* gW = Wd + (size_t)(col0 + cn) * K + ck;
  const _Float16* A  = Hrot + (size_t)row0 * K;

  v8f acc[4] = {};
  const int NK = K / 32;

#if defined(USE_ASYNC_COPY)
  async_copy16(gW, &sW[0][cn * 32 + ck]);
#else
  v8h rw = *(const v8h*)(gW);
#endif

  for (int k = 0; k < NK - 1; ++k) {
    const int cur = k & 1, nxt = cur ^ 1;
    const int kk = k * 32;
#if defined(USE_ASYNC_COPY)
    async_copy16(gW + (size_t)(k + 1) * 32, &sW[nxt][cn * 32 + ck]);
    __builtin_amdgcn_s_wait_asynccnt(1);
    __syncthreads();
#else
    *(v8h*)(&sW[cur][cn * 32 + ck]) = rw;
    __syncthreads();
    rw = *(const v8h*)(gW + (size_t)(k + 1) * 32);
#endif
    __builtin_prefetch(A + kk + 128, 0, 3);
    v16h a = load_frag(A + kk, K, lane);
    #pragma unroll
    for (int t = 0; t < 4; ++t) {
      v16h b = load_frag_lds(&sW[cur][(t * 16) * 32], lane);
      acc[t] = wmma_f16(a, b, acc[t]);
    }
    __syncthreads();
  }
  {
    const int cur = (NK - 1) & 1;
    const int kk = (NK - 1) * 32;
#if defined(USE_ASYNC_COPY)
    __builtin_amdgcn_s_wait_asynccnt(0);
    __syncthreads();
#else
    *(v8h*)(&sW[cur][cn * 32 + ck]) = rw;
    __syncthreads();
#endif
    v16h a = load_frag(A + kk, K, lane);
    #pragma unroll
    for (int t = 0; t < 4; ++t) {
      v16h b = load_frag_lds(&sW[cur][(t * 16) * 32], lane);
      acc[t] = wmma_f16(a, b, acc[t]);
    }
  }

  const int rbase = (lane >> 4) << 3;
  const int c     = lane & 15;
  #pragma unroll
  for (int t = 0; t < 4; ++t) {
    #pragma unroll
    for (int i = 0; i < 8; ++i) {
      const int n = col0 + t * 16 + c;
      out[(size_t)(row0 + rbase + i) * N + n] = acc[t][i] + bias[n];
    }
  }
}

// ---------------------------------------------------------------------------
// Launch
// ---------------------------------------------------------------------------
extern "C" void kernel_launch(void* const* d_in, const int* in_sizes, int n_in,
                              void* d_out, int out_size, void* d_ws, size_t ws_size,
                              hipStream_t stream) {
  const float* x           = (const float*)d_in[0];
  const float* codebook    = (const float*)d_in[1];
  const float* gate_norms  = (const float*)d_in[2];
  const float* gate_bias   = (const float*)d_in[3];
  const float* up_norms    = (const float*)d_in[4];
  const float* up_bias     = (const float*)d_in[5];
  const float* down_norms  = (const float*)d_in[6];
  const float* down_bias   = (const float*)d_in[7];
  const float* rot_gate    = (const float*)d_in[8];
  const float* rot_up      = (const float*)d_in[9];
  const float* rot_down    = (const float*)d_in[10];
  const int*   gate_packed = (const int*)d_in[11];
  const int*   up_packed   = (const int*)d_in[12];
  const int*   down_packed = (const int*)d_in[13];

  // Workspace layout (bytes). Total 136 MB.
  char* ws = (char*)d_ws;
  const size_t MB = 1024ull * 1024ull;
  _Float16* Wg   = (_Float16*)(ws);                  // 32 MB  [INTER, IN]
  _Float16* Wu   = (_Float16*)(ws + 32 * MB);        // 32 MB  [INTER, IN]
  _Float16* Wd   = (_Float16*)(ws + 64 * MB);        // 32 MB  [IN, INTER]
  _Float16* Xg   = (_Float16*)(ws + 96 * MB);        // 4  MB  [B, IN]
  _Float16* Xu   = (_Float16*)(ws + 100 * MB);       // 4  MB  [B, IN]
  _Float16* H    = (_Float16*)(ws + 104 * MB);       // 16 MB  [B, INTER]
  _Float16* Hrot = (_Float16*)(ws + 120 * MB);       // 16 MB  [B, INTER]

  {
    const int total = INTER_DIM * (IN_DIM / 2);      // 8,388,608
    const int blocks = (total + 255) / 256;
    dequant_kernel<<<blocks, 256, 0, stream>>>(gate_packed, codebook, gate_norms,
                                               Wg, INTER_DIM, IN_DIM);
    dequant_kernel<<<blocks, 256, 0, stream>>>(up_packed, codebook, up_norms,
                                               Wu, INTER_DIM, IN_DIM);
    const int totald = IN_DIM * (INTER_DIM / 2);
    dequant_kernel<<<(totald + 255) / 256, 256, 0, stream>>>(
        down_packed, codebook, down_norms, Wd, IN_DIM, INTER_DIM);
  }
  {
    dim3 grid(BATCH / 128, IN_DIM / GRP);            // (8, 16)
    rotate_kernel<float><<<grid, 256, 0, stream>>>(x, rot_gate, Xg, IN_DIM);
    rotate_kernel<float><<<grid, 256, 0, stream>>>(x, rot_up,   Xu, IN_DIM);
  }
  {
    dim3 grid(BATCH / 128, INTER_DIM / 64);          // (8, 128)
    gateup_kernel<<<grid, 256, 0, stream>>>(Xg, Xu, Wg, Wu, gate_bias, up_bias,
                                            H, IN_DIM, INTER_DIM);
  }
  {
    dim3 grid(BATCH / 128, INTER_DIM / GRP);         // (8, 64)
    rotate_kernel<_Float16><<<grid, 256, 0, stream>>>(H, rot_down, Hrot, INTER_DIM);
  }
  {
    dim3 grid(BATCH / 128, IN_DIM / 64);             // (8, 32)
    down_kernel<<<grid, 256, 0, stream>>>(Hrot, Wd, down_bias, (float*)d_out,
                                          INTER_DIM, IN_DIM);
  }
}
